// BaichuanAttention_549755814458
// MI455X (gfx1250) — matrix operand order repaired and checked by reading, not compiled, not run on hardware
//
#include <hip/hip_runtime.h>
#include <hip/hip_bf16.h>

typedef __attribute__((ext_vector_type(16))) _Float16 v16h;
typedef __attribute__((ext_vector_type(8)))  float    v8f;

union HF { v16h v; unsigned int u[8]; _Float16 h[16]; };

#define WMMA_F16(a, b, c) \
  __builtin_amdgcn_wmma_f32_16x16x32_f16(false, (a), false, (b), (short)0, (c), false, false)

static __device__ inline v8f vzero8() {
  v8f z;
#pragma unroll
  for (int j = 0; j < 8; ++j) z[j] = 0.0f;
  return z;
}

// ---------------------------------------------------------------------------
// Kernel 1: fp32 -> fp16 conversion (hidden_states -> Xh), 8B stores
// ---------------------------------------------------------------------------
__global__ void cvt_f32_to_f16(const float* __restrict__ x, _Float16* __restrict__ y,
                               const int n4) {
  const int i = blockIdx.x * blockDim.x + threadIdx.x;
  if (i >= n4) return;
  const float4 f = ((const float4*)x)[i];
  union { _Float16 h[4]; unsigned long long q; } p;
  p.h[0] = (_Float16)f.x; p.h[1] = (_Float16)f.y;
  p.h[2] = (_Float16)f.z; p.h[3] = (_Float16)f.w;
  ((unsigned long long*)y)[i] = p.q;
}

// ---------------------------------------------------------------------------
// Kernel 2/4: streaming GEMM  C[512][Ntot] = A_f16[512][4096] @ B_f32[4096][Ntot]
// Grid = Ntot/64 WGs (weights streamed from HBM exactly once).
// Block = 512 threads = 16 waves; wave w owns rows 32w..32w+31, all 64 cols.
// MODE 0: scatter f16 to per-head q/k/v layout [B,H,64,128].
// MODE 1: write f32 rows to d_out [512][4096].
// ---------------------------------------------------------------------------
template<int MODE>
__global__ __launch_bounds__(512)
void gemm_kernel(const _Float16* __restrict__ A, const float* __restrict__ B,
                 const int Ntot,
                 _Float16* __restrict__ qh, _Float16* __restrict__ kh,
                 _Float16* __restrict__ vh, float* __restrict__ outf) {
  const int c0   = blockIdx.x * 64;
  const int tid  = threadIdx.x;
  const int wave = tid >> 5, lane = tid & 31;
  const int lm   = lane & 15, lg = lane >> 4;

  __shared__ __align__(16) _Float16 Al[512 * 40];  // [row][k] pad 40 (80B rows, conflict-free frags)
  __shared__ __align__(16) _Float16 Bl[64 * 34];   // [n][k] transposed, pad 34

  v8f acc[2][4];
#pragma unroll
  for (int mt = 0; mt < 2; ++mt)
#pragma unroll
    for (int nt = 0; nt < 4; ++nt) acc[mt][nt] = vzero8();

  const int kB = tid >> 4, nB = (tid & 15) * 4;

  for (int kk = 0; kk < 4096; kk += 32) {
    __syncthreads();
    {  // A chunk: 512 rows x 32 halves, one row per thread, 4x16B
      const uint4* s = (const uint4*)(A + (size_t)tid * 4096 + kk);
      uint4*       d = (uint4*)(Al + tid * 40);
      d[0] = s[0]; d[1] = s[1]; d[2] = s[2]; d[3] = s[3];
    }
    {  // B chunk: 32 x 64 f32, coalesced float4, transposed f16 into LDS
      const float4 f = *(const float4*)(B + (size_t)(kk + kB) * Ntot + c0 + nB);
      Bl[(nB + 0) * 34 + kB] = (_Float16)f.x;
      Bl[(nB + 1) * 34 + kB] = (_Float16)f.y;
      Bl[(nB + 2) * 34 + kB] = (_Float16)f.z;
      Bl[(nB + 3) * 34 + kB] = (_Float16)f.w;
    }
    // prefetch next K-step of the weight stream (indirect-free, keeps HBM pipe full)
    if (kk + 32 < 4096)
      __builtin_prefetch(B + (size_t)(kk + 32 + kB) * Ntot + c0 + nB, 0, 0);
    __syncthreads();

    HF af0, af1;
#pragma unroll
    for (int r = 0; r < 8; ++r) {
      const int kof = 16 * (r >> 2) + 8 * lg + 2 * (r & 3);
      af0.u[r] = *(const unsigned int*)&Al[(wave * 32 + lm) * 40 + kof];
      af1.u[r] = *(const unsigned int*)&Al[(wave * 32 + 16 + lm) * 40 + kof];
    }
#pragma unroll
    for (int nt = 0; nt < 4; ++nt) {
      HF bf;
#pragma unroll
      for (int r = 0; r < 8; ++r)
        bf.u[r] = *(const unsigned int*)&Bl[(nt * 16 + lm) * 34 + 16 * (r >> 2) + 8 * lg + 2 * (r & 3)];
      acc[0][nt] = WMMA_F16(af0.v, bf.v, acc[0][nt]);
      acc[1][nt] = WMMA_F16(af1.v, bf.v, acc[1][nt]);
    }
  }

#pragma unroll
  for (int mt = 0; mt < 2; ++mt)
#pragma unroll
    for (int nt = 0; nt < 4; ++nt)
#pragma unroll
      for (int j = 0; j < 8; ++j) {
        const int r = wave * 32 + mt * 16 + j + 8 * lg;   // 0..511
        const int c = c0 + nt * 16 + lm;
        const float val = acc[mt][nt][j];
        if (MODE == 0) {
          const int part = c >> 12;          // 0=q,1=k,2=v
          const int cc   = c & 4095;
          const int h    = cc >> 7, d = cc & 127;
          _Float16* dst  = (part == 0) ? qh : (part == 1) ? kh : vh;
          dst[((((size_t)(r >> 6)) * 32 + h) * 64 + (r & 63)) * 128 + d] = (_Float16)val;
        } else {
          outf[(size_t)r * 4096 + c] = val;
        }
      }
}

// ---------------------------------------------------------------------------
// Kernel 3: flash attention with ALiBi + causal mask + paged KV gather.
// One WG per (b,h): 128 threads = 4 waves; wave w owns q rows 16w..16w+15.
// New tokens (pos in [hist, hist+64)) come from kh/vh instead of cache
// (equivalent to the reference's cache scatter, without mutating inputs).
// Next physical block is software-prefetched: the block_offsets indirection
// defeats any hardware streamer, and this gather is the dominant HBM traffic.
// ---------------------------------------------------------------------------
__global__ __launch_bounds__(128)
void attn_kernel(const _Float16* __restrict__ qh, const _Float16* __restrict__ kh,
                 const _Float16* __restrict__ vh,
                 const float* __restrict__ k_cache, const float* __restrict__ v_cache,
                 const long long* __restrict__ block_offsets,
                 const int* __restrict__ histp,
                 _Float16* __restrict__ attn_out) {
  const int bh   = blockIdx.x;
  const int b    = bh >> 5, h = bh & 31;
  const int tid  = threadIdx.x;
  const int wave = tid >> 5, lane = tid & 31;
  const int lm   = lane & 15, lg = lane >> 4;
  const int hist = histp[0];

  __shared__ __align__(16) _Float16 kt[64 * 136];   // K chunk [slot][d], pad 136
  __shared__ __align__(16) _Float16 vt[128 * 66];   // V chunk transposed [d][slot], pad 66
  __shared__ __align__(16) _Float16 pl[4][16 * 72]; // per-wave P tile [row][slot], pad 72

  const float scale = 0.08838834764831845f;                 // 1/sqrt(128)
  const float slope = exp2f(-0.25f * (float)(h + 1));       // 2^(-8(h+1)/32)

  // Q A-fragments (16 rows x 128 dims -> 4 K-chunks), resident all kernel
  HF aQ[4];
  {
    const _Float16* qbase = qh + (((size_t)b * 32 + h) * 64 + (wave * 16 + lm)) * 128;
#pragma unroll
    for (int kc = 0; kc < 4; ++kc)
#pragma unroll
      for (int r = 0; r < 8; ++r) {
        const int d = kc * 32 + 16 * (r >> 2) + 8 * lg + 2 * (r & 3);
        aQ[kc].u[r] = *(const unsigned int*)&qbase[d];
      }
  }

  v8f O[8];
#pragma unroll
  for (int dt = 0; dt < 8; ++dt) O[dt] = vzero8();
  float m_run[8], l_run[8];
#pragma unroll
  for (int j = 0; j < 8; ++j) { m_run[j] = -1e30f; l_run[j] = 0.0f; }

  for (int bb = 0; bb < 64; ++bb) {
    const size_t blk = (size_t)block_offsets[(size_t)b * 64 + bb];
    __syncthreads();
    // cooperative KV chunk load: 64 slots x 128 dims, d = tid (coalesced)
#pragma unroll 4
    for (int i = 0; i < 64; ++i) {
      const int slot = i, d = tid;
      const int p = bb * 64 + slot;
      _Float16 kvh, vvh;
      if ((p - hist) >= 0 && (p - hist) < 64) {
        const size_t nb = (((size_t)b * 32 + h) * 64 + (p - hist)) * 128 + d;
        kvh = kh[nb]; vvh = vh[nb];
      } else {
        const size_t cb = ((blk * 64 + slot) * 32 + h) * 128 + d;
        kvh = (_Float16)k_cache[cb];
        vvh = (_Float16)v_cache[cb];
      }
      kt[slot * 136 + d] = kvh;
      vt[d * 66 + slot]  = vvh;
    }
    // prefetch next block (this head's 64 rows of 512B in K and V cache)
    if (bb + 1 < 64) {
      const size_t nblk = (size_t)block_offsets[(size_t)b * 64 + bb + 1];
      const int slot = tid & 63;                 // threads 0-63: K, 64-127: V
      const float* src = (tid < 64) ? k_cache : v_cache;
      const float* row = src + ((nblk * 64 + slot) * 32 + h) * 128;
#pragma unroll
      for (int o = 0; o < 4; ++o)
        __builtin_prefetch(row + o * 32, 0, 0);  // 4 x 128B cachelines
    }
    __syncthreads();

    // S = scale * Q K^T + alibi, causal-masked
    float sv[4][8];
#pragma unroll
    for (int nt = 0; nt < 4; ++nt) {
      v8f sacc = vzero8();
#pragma unroll
      for (int kc = 0; kc < 4; ++kc) {
        HF bk;
#pragma unroll
        for (int r = 0; r < 8; ++r)
          bk.u[r] = *(const unsigned int*)&kt[(nt * 16 + lm) * 136 + kc * 32 + 16 * (r >> 2) + 8 * lg + 2 * (r & 3)];
        sacc = WMMA_F16(aQ[kc].v, bk.v, sacc);
      }
      const int p = bb * 64 + nt * 16 + lm;
#pragma unroll
      for (int j = 0; j < 8; ++j) {
        const int qp = hist + wave * 16 + j + 8 * lg;
        const float s = sacc[j] * scale - slope * (float)(qp - p);
        sv[nt][j] = (p > qp) ? -1e30f : s;
      }
    }

    // online softmax: row max via 16-lane butterfly (wave32 groups)
    float mn[8];
#pragma unroll
    for (int j = 0; j < 8; ++j) {
      float m = fmaxf(fmaxf(sv[0][j], sv[1][j]), fmaxf(sv[2][j], sv[3][j]));
      m = fmaxf(m, __shfl_xor(m, 1, 32));
      m = fmaxf(m, __shfl_xor(m, 2, 32));
      m = fmaxf(m, __shfl_xor(m, 4, 32));
      m = fmaxf(m, __shfl_xor(m, 8, 32));
      mn[j] = fmaxf(m, m_run[j]);
    }
    float rs[8];
#pragma unroll
    for (int j = 0; j < 8; ++j) rs[j] = 0.0f;
#pragma unroll
    for (int nt = 0; nt < 4; ++nt)
#pragma unroll
      for (int j = 0; j < 8; ++j) {
        const float pv = __expf(sv[nt][j] - mn[j]);
        rs[j] += pv;
        pl[wave][(j + 8 * lg) * 72 + nt * 16 + lm] = (_Float16)pv;  // wave-private
      }
#pragma unroll
    for (int j = 0; j < 8; ++j) {
      float r = rs[j];
      r += __shfl_xor(r, 1, 32);
      r += __shfl_xor(r, 2, 32);
      r += __shfl_xor(r, 4, 32);
      r += __shfl_xor(r, 8, 32);
      const float alpha = __expf(m_run[j] - mn[j]);
      l_run[j] = l_run[j] * alpha + r;
      m_run[j] = mn[j];
#pragma unroll
      for (int dt = 0; dt < 8; ++dt) O[dt][j] *= alpha;
    }

    // O += P @ V  (P re-read from wave-private LDS as A-fragments)
#pragma unroll
    for (int kc = 0; kc < 2; ++kc) {
      HF aP;
#pragma unroll
      for (int r = 0; r < 8; ++r)
        aP.u[r] = *(const unsigned int*)
            &pl[wave][lm * 72 + kc * 32 + 16 * (r >> 2) + 8 * lg + 2 * (r & 3)];
#pragma unroll
      for (int dt = 0; dt < 8; ++dt) {
        HF bV;
#pragma unroll
        for (int r = 0; r < 8; ++r)
          bV.u[r] = *(const unsigned int*)&vt[(dt * 16 + lm) * 66 + kc * 32 + 16 * (r >> 2) + 8 * lg + 2 * (r & 3)];
        O[dt] = WMMA_F16(aP.v, bV.v, O[dt]);
      }
    }
  }

  // normalize and emit f16 rows [512][4096] for the output projection
#pragma unroll
  for (int j = 0; j < 8; ++j) {
    const float inv = 1.0f / l_run[j];
    const int row = b * 64 + wave * 16 + j + 8 * lg;
#pragma unroll
    for (int dt = 0; dt < 8; ++dt)
      attn_out[(size_t)row * 4096 + h * 128 + dt * 16 + lm] = (_Float16)(O[dt][j] * inv);
  }
}

// ---------------------------------------------------------------------------
extern "C" void kernel_launch(void* const* d_in, const int* in_sizes, int n_in,
                              void* d_out, int out_size, void* d_ws, size_t ws_size,
                              hipStream_t stream) {
  (void)in_sizes; (void)n_in; (void)out_size; (void)ws_size;
  const float*     hs   = (const float*)d_in[0];
  const float*     Wqkv = (const float*)d_in[1];
  const float*     Wo   = (const float*)d_in[2];
  const float*     kc   = (const float*)d_in[3];
  const float*     vc   = (const float*)d_in[4];
  const long long* bo   = (const long long*)d_in[5];   // int64 per reference
  const int*       hp   = (const int*)d_in[6];         // LE low word OK for i32/i64
  float*           out  = (float*)d_out;

  const size_t NTOK = 512, HID = 4096, HEADS = 32, QL = 64, D = 128, BATCH = 8;
  _Float16* Xh = (_Float16*)d_ws;                         // 512x4096 f16
  _Float16* qh = Xh + NTOK * HID;                         // [B,H,64,128] f16
  _Float16* kh = qh + BATCH * HEADS * QL * D;
  _Float16* vh = kh + BATCH * HEADS * QL * D;
  _Float16* at = vh + BATCH * HEADS * QL * D;             // 512x4096 f16

  const int n4 = (int)(NTOK * HID / 4);
  cvt_f32_to_f16<<<(n4 + 255) / 256, 256, 0, stream>>>(hs, Xh, n4);
  gemm_kernel<0><<<12288 / 64, 512, 0, stream>>>(Xh, Wqkv, 12288, qh, kh, vh, nullptr);
  attn_kernel<<<BATCH * HEADS, 128, 0, stream>>>(qh, kh, vh, kc, vc, bo, hp, at);
  gemm_kernel<1><<<4096 / 64, 512, 0, stream>>>(at, Wo, 4096, nullptr, nullptr, nullptr, out);
}
